// RotomerLdq_36352603194129
// MI455X (gfx1250) — matrix operand order repaired and checked by reading, not compiled, not run on hardware
//
#include <hip/hip_runtime.h>

#define NN      20000
#define FEATD   128
#define NCOLS   256
#define CALD    16
#define NGRAPH  8
#define E_CHAIN 20000
#define E_RAD   320000
#define E_ALL   (E_CHAIN + E_RAD + NN)   /* 360000: edge_index ++ radius ++ self loops */
#define E_OFF1  E_CHAIN                  /* layers 1,2 use radius ++ self loops */
#define E_LATE  (E_RAD + NN)             /* 340000 */

typedef __attribute__((ext_vector_type(16))) __bf16 v16bf;
typedef __attribute__((ext_vector_type(8)))  __bf16 v8bf;
typedef __attribute__((ext_vector_type(8)))  float  v8f;

union ABfrag { v16bf v; v8bf h[2]; };

// ---- monotone float<->uint mapping for atomic max over signed floats ----
static __device__ __forceinline__ unsigned fmap_u(float x) {
  unsigned u = __float_as_uint(x);
  return (u & 0x80000000u) ? ~u : (u | 0x80000000u);
}
static __device__ __forceinline__ float funmap_u(unsigned m) {
  return __uint_as_float((m & 0x80000000u) ? (m ^ 0x80000000u) : ~m);
}
static __device__ __forceinline__ float hswish(float v) {
  float r = v + 3.0f; r = r < 0.0f ? 0.0f : (r > 6.0f ? 6.0f : r);
  return v * r * (1.0f / 6.0f);
}

// ---------------- node embedding: concat(tables, pos @ xyz_W + b) ----------------
__global__ __launch_bounds__(256) void k_embed(const int* __restrict__ xi,
    const float* __restrict__ pos, const float* __restrict__ rt,
    const float* __restrict__ at, const float* __restrict__ ct,
    const float* __restrict__ xw, const float* __restrict__ xb,
    float* __restrict__ hC, float* __restrict__ hB, float* __restrict__ hA) {
  int tid = blockIdx.x * 256 + threadIdx.x;
  if (tid >= NN * FEATD) return;
  int n = tid >> 7, f = tid & 127;
  float v;
  if (f < 16)       v = rt[xi[n*3+0]*16 + f];
  else if (f < 20)  v = at[xi[n*3+1]*4 + (f-16)];
  else if (f < 36)  v = ct[xi[n*3+2]*16 + (f-20)];
  else {
    int j = f - 36;
    v = pos[n*3+0]*xw[j] + pos[n*3+1]*xw[92+j] + pos[n*3+2]*xw[184+j] + xb[j];
  }
  hC[tid] = v; hB[tid] = v; hA[tid] = v;
}

// ---------------- build combined edge list: [chain | radius | self] ----------------
__global__ __launch_bounds__(256) void k_edges(const int* __restrict__ ei,
    const int* __restrict__ ri, int* __restrict__ es, int* __restrict__ ed) {
  int i = blockIdx.x * 256 + threadIdx.x;
  if (i >= E_ALL) return;
  if (i < E_CHAIN)                { es[i] = ei[i];            ed[i] = ei[E_CHAIN + i]; }
  else if (i < E_CHAIN + E_RAD)   { int j = i - E_CHAIN; es[i] = ri[j]; ed[i] = ri[E_RAD + j]; }
  else                            { int n = i - (E_CHAIN + E_RAD); es[i] = n; ed[i] = n; }
}

// ---------------- f32 -> bf16 ----------------
__global__ __launch_bounds__(256) void k_cvt(const float* __restrict__ s,
                                             __bf16* __restrict__ d, int n) {
  int i = blockIdx.x * 256 + threadIdx.x;
  if (i < n) d[i] = (__bf16)s[i];
}

// ---------------- weight transpose+convert: W[K=128, ncols] -> Wt[ncols][128] ----------------
__global__ __launch_bounds__(256) void k_wprep(const float* __restrict__ W,
                                               __bf16* __restrict__ Wt, int ncols) {
  int i = blockIdx.x * 256 + threadIdx.x;
  if (i >= ncols * FEATD) return;
  int c = i >> 7, k = i & 127;
  Wt[c * FEATD + k] = (__bf16)W[k * ncols + c];
}

// ---------------- WMMA GEMM: C[M,COLS] = A[M,128] x Wt^T ----------------
// COLS compile-time -> constant stride stores (single s_clause of b32 stores).
// Each wave: one 16-row A fragment set reused across CPW adjacent column tiles.
template<int COLS, int CPW>
__global__ __launch_bounds__(256) void k_gemm(const __bf16* __restrict__ A,
    const __bf16* __restrict__ Bt, float* __restrict__ C, int M) {
  constexpr int colTn  = COLS / 16;
  constexpr int groups = colTn / CPW;
  int wave = (blockIdx.x * 256 + (int)threadIdx.x) >> 5;
  int lane = threadIdx.x & 31;
  int tiles = (M >> 4) * groups;
  if (wave >= tiles) return;
  int rowT = wave / groups, colG = wave - rowT * groups;
  int lo = lane & 15, hi = lane >> 4;

  // A 16x32 bf16 fragments for all 4 K-chunks, loaded once per wave.
  const __bf16* ar = A + (size_t)(rowT * 16 + lo) * FEATD;
  ABfrag av[4];
#pragma unroll
  for (int kc = 0; kc < 4; ++kc) {
    int kb = kc * 32;
    av[kc].h[0] = *(const v8bf*)(ar + kb + hi * 8);        // K = kb + hi*8 + j
    av[kc].h[1] = *(const v8bf*)(ar + kb + 16 + hi * 8);   // K = kb + 16 + hi*8 + j
  }

  v8f acc[CPW];
#pragma unroll
  for (int c = 0; c < CPW; ++c) acc[c] = (v8f){};

#pragma unroll
  for (int c = 0; c < CPW; ++c) {
    const __bf16* br = Bt + (size_t)((colG * CPW + c) * 16 + lo) * FEATD;
    ABfrag bv[4];
#pragma unroll
    for (int kc = 0; kc < 4; ++kc) {
      int kb = kc * 32;
      bv[kc].h[0] = *(const v8bf*)(br + kb + hi * 8);       // K = kb + hi*8 + i
      bv[kc].h[1] = *(const v8bf*)(br + kb + 16 + hi * 8);
    }
#pragma unroll
    for (int kc = 0; kc < 4; ++kc)
      acc[c] = __builtin_amdgcn_wmma_f32_16x16x32_bf16(false, av[kc].v, false, bv[kc].v,
                                                       (short)0, acc[c], false, false);
  }

  // C/D layout: VGPR r -> row M = r + hi*8, col N = lo; constant stride COLS.
#pragma unroll
  for (int c = 0; c < CPW; ++c) {
    float* cr = C + (size_t)(rowT * 16) * COLS + (colG * CPW + c) * 16 + lo;
#pragma unroll
    for (int r = 0; r < 8; ++r) cr[(r + hi * 8) * COLS] = acc[c][r];
  }
}

// ---------------- attention coefficients: one wave per node, shfl reduction ----------------
__global__ __launch_bounds__(256) void k_att(const float* __restrict__ hh,
    const float* __restrict__ aL, float* __restrict__ als, float* __restrict__ ald) {
  int wave = (blockIdx.x * 256 + (int)threadIdx.x) >> 5;
  int lane = threadIdx.x & 31;
  if (wave >= NN) return;
  const float4* row = (const float4*)(hh + (size_t)wave * NCOLS);
  const float4* as4 = (const float4*)aL;           // a[0], both heads: 64 float4
  const float4* ad4 = (const float4*)(aL + 256);   // a[1]
  float4 x0 = row[lane], x1 = row[lane + 32];      // head0 / head1 slices
  float4 s0 = as4[lane], s1 = as4[lane + 32];
  float4 d0 = ad4[lane], d1 = ad4[lane + 32];
  float ss0 = x0.x*s0.x + x0.y*s0.y + x0.z*s0.z + x0.w*s0.w;
  float ss1 = x1.x*s1.x + x1.y*s1.y + x1.z*s1.z + x1.w*s1.w;
  float sd0 = x0.x*d0.x + x0.y*d0.y + x0.z*d0.z + x0.w*d0.w;
  float sd1 = x1.x*d1.x + x1.y*d1.y + x1.z*d1.z + x1.w*d1.w;
#pragma unroll
  for (int m = 16; m; m >>= 1) {
    ss0 += __shfl_xor(ss0, m, 32); ss1 += __shfl_xor(ss1, m, 32);
    sd0 += __shfl_xor(sd0, m, 32); sd1 += __shfl_xor(sd1, m, 32);
  }
  if (lane == 0) {
    als[wave*2+0] = ss0; als[wave*2+1] = ss1;
    ald[wave*2+0] = sd0; ald[wave*2+1] = sd1;
  }
}

// ---------------- per-layer accumulator init ----------------
__global__ __launch_bounds__(256) void k_init(float* __restrict__ outb,
    unsigned* __restrict__ mu, float* __restrict__ den) {
  int i = blockIdx.x * 256 + threadIdx.x;
  if (i < NN * FEATD) outb[i] = 0.f;
  if (i < NN * 2) { mu[i] = 0u; den[i] = 0.f; }
}

// ---------------- softmax pass 1: segment max (mapped uint atomicMax) ----------------
__global__ __launch_bounds__(256) void k_p1(const int* __restrict__ es,
    const int* __restrict__ ed, const float* __restrict__ als,
    const float* __restrict__ ald, unsigned* __restrict__ mu, int E) {
  int e = blockIdx.x * 256 + threadIdx.x;
  if (e >= E) return;
  int s = es[e], d = ed[e];
#pragma unroll
  for (int h = 0; h < 2; ++h) {
    float v = als[s*2+h] + ald[d*2+h];
    v = v > 0.f ? v : 0.2f * v;
    atomicMax(&mu[d*2+h], fmap_u(v));
  }
}

// ---------------- softmax pass 2: exp + segment sum ----------------
__global__ __launch_bounds__(256) void k_p2(const int* __restrict__ es,
    const int* __restrict__ ed, const float* __restrict__ als,
    const float* __restrict__ ald, const unsigned* __restrict__ mu,
    float* __restrict__ den, float* __restrict__ exb, int E) {
  int e = blockIdx.x * 256 + threadIdx.x;
  if (e >= E) return;
  int s = es[e], d = ed[e];
#pragma unroll
  for (int h = 0; h < 2; ++h) {
    float v = als[s*2+h] + ald[d*2+h];
    v = v > 0.f ? v : 0.2f * v;
    float ev = __expf(v - funmap_u(mu[d*2+h]));
    exb[e*2+h] = ev;
    atomicAdd(&den[d*2+h], ev);
  }
}

// ---------------- softmax pass 3: scatter alpha * hh[src], mean over heads ----------------
__global__ __launch_bounds__(256) void k_p3(const int* __restrict__ es,
    const int* __restrict__ ed, const float* __restrict__ exb,
    const float* __restrict__ den, const float* __restrict__ hh,
    float* __restrict__ outb, int E) {
  int t = blockIdx.x * 256 + threadIdx.x;
  int e = t >> 2, q = t & 3;
  if (e >= E) return;
  int s = es[e], d = ed[e];
  float a0 = exb[e*2+0] / den[d*2+0];
  float a1 = exb[e*2+1] / den[d*2+1];
  const float4* h0 = (const float4*)(hh + (size_t)s * NCOLS + q * 32);
  const float4* h1 = (const float4*)(hh + (size_t)s * NCOLS + 128 + q * 32);
  float* od = outb + (size_t)d * FEATD + q * 32;
#pragma unroll
  for (int i = 0; i < 8; ++i) {
    float4 x0 = h0[i], x1 = h1[i];
    atomicAdd(&od[i*4+0], 0.5f * (a0 * x0.x + a1 * x1.x));
    atomicAdd(&od[i*4+1], 0.5f * (a0 * x0.y + a1 * x1.y));
    atomicAdd(&od[i*4+2], 0.5f * (a0 * x0.z + a1 * x1.z));
    atomicAdd(&od[i*4+3], 0.5f * (a0 * x0.w + a1 * x1.w));
  }
}

// ---------------- bias + optional hard_swish -> next layer features ----------------
__global__ __launch_bounds__(256) void k_bact(const float* __restrict__ outb,
    const float* __restrict__ b, float* __restrict__ hnew, int act) {
  int tid = blockIdx.x * 256 + threadIdx.x;
  if (tid >= NN * FEATD) return;
  float v = outb[tid] + b[tid & 127];
  hnew[tid] = act ? hswish(v) : v;
}

// ---------------- out-proj bias + optional activation (in-place, [N,16]) ----------------
__global__ __launch_bounds__(256) void k_obact(float* __restrict__ p,
    const float* __restrict__ b, int act) {
  int tid = blockIdx.x * 256 + threadIdx.x;
  if (tid >= NN * CALD) return;
  float v = p[tid] + b[tid & 15];
  p[tid] = act ? hswish(v) : v;
}

__global__ void k_zero8(float* __restrict__ o) { if (threadIdx.x < NGRAPH) o[threadIdx.x] = 0.f; }

// ---------------- per-edge energy + per-graph reduction ----------------
__global__ __launch_bounds__(256) void k_energy(const int* __restrict__ ri,
    const float* __restrict__ pos, const int* __restrict__ batch,
    const float* __restrict__ Cf, const float* __restrict__ Bf,
    const float* __restrict__ Af, float* __restrict__ out) {
  __shared__ float sacc[NGRAPH];
  if (threadIdx.x < NGRAPH) sacc[threadIdx.x] = 0.f;
  __syncthreads();
  int e = blockIdx.x * 256 + threadIdx.x;
  if (e < E_RAD) {
    int r0 = ri[e], r1 = ri[E_RAD + e];
    float dx = pos[r0*3+0] - pos[r1*3+0];
    float dy = pos[r0*3+1] - pos[r1*3+1];
    float dz = pos[r0*3+2] - pos[r1*3+2];
    float dist = dx*dx + dy*dy + dz*dz;   // squared distance
    const float* c0 = Cf + r0*CALD; const float* c1 = Cf + r1*CALD;
    const float* b0 = Bf + r0*CALD; const float* b1 = Bf + r1*CALD;
    const float* a0 = Af + r0*CALD; const float* a1 = Af + r1*CALD;
    float sum = 0.f, p = 1.f;
#pragma unroll
    for (int f = 0; f < CALD; ++f) {
      float Cp = 0.5f * (c0[f] + c1[f]);
      float Bp = 0.5f * (b0[f] + b1[f]);
      float Ap = 0.5f * (a0[f] + a1[f]);
      sum += Cp * (1.0f / p + Bp) + Ap;
      p *= dist;
    }
    atomicAdd(&sacc[batch[r0]], sum * (1.0f / (float)CALD));
  }
  __syncthreads();
  if (threadIdx.x < NGRAPH) atomicAdd(&out[threadIdx.x], sacc[threadIdx.x]);
}

extern "C" void kernel_launch(void* const* d_in, const int* in_sizes, int n_in,
                              void* d_out, int out_size, void* d_ws, size_t ws_size,
                              hipStream_t stream) {
  const int*   x_idx = (const int*)  d_in[0];
  const float* pos   = (const float*)d_in[1];
  const int*   ei    = (const int*)  d_in[2];
  const int*   ri    = (const int*)  d_in[3];
  const int*   batch = (const int*)  d_in[4];
  const float* res_t = (const float*)d_in[5];
  const float* atm_t = (const float*)d_in[6];
  const float* cnt_t = (const float*)d_in[7];
  const float* xyzW  = (const float*)d_in[8];
  const float* xyzb  = (const float*)d_in[9];
  const float* Wb_[3]  = {(const float*)d_in[10], (const float*)d_in[13], (const float*)d_in[16]};
  const float* ab_[3]  = {(const float*)d_in[11], (const float*)d_in[14], (const float*)d_in[17]};
  const float* bb_[3]  = {(const float*)d_in[12], (const float*)d_in[15], (const float*)d_in[18]};
  const float* Wo_[3]  = {(const float*)d_in[19], (const float*)d_in[21], (const float*)d_in[23]};
  const float* bo_[3]  = {(const float*)d_in[20], (const float*)d_in[22], (const float*)d_in[24]};
  float* out = (float*)d_out;

  // ---- workspace carve-up (~76 MB, 256B aligned) ----
  char* base = (char*)d_ws;
  size_t off = 0;
  auto alloc = [&](size_t bytes) -> char* {
    char* p = base + off; off = (off + bytes + 255) & ~(size_t)255; return p;
  };
  float*    hbr[3]; // per-branch node features (C, B, A)
  for (int b = 0; b < 3; ++b) hbr[b] = (float*)alloc((size_t)NN * FEATD * 4);
  float*    hh   = (float*)   alloc((size_t)NN * NCOLS * 4);
  float*    outb = (float*)   alloc((size_t)NN * FEATD * 4);
  __bf16*   hbf  = (__bf16*)  alloc((size_t)NN * FEATD * 2);
  __bf16*   Wt   = (__bf16*)  alloc((size_t)NCOLS * FEATD * 2);
  float*    als  = (float*)   alloc((size_t)NN * 2 * 4);
  float*    ald  = (float*)   alloc((size_t)NN * 2 * 4);
  unsigned* mu   = (unsigned*)alloc((size_t)NN * 2 * 4);
  float*    den  = (float*)   alloc((size_t)NN * 2 * 4);
  float*    exb  = (float*)   alloc((size_t)E_ALL * 2 * 4);
  int*      esrc = (int*)     alloc((size_t)E_ALL * 4);
  int*      edst = (int*)     alloc((size_t)E_ALL * 4);
  float*    fin[3];
  for (int b = 0; b < 3; ++b) fin[b] = (float*)alloc((size_t)NN * CALD * 4);

  auto grid = [](long n) { return dim3((unsigned)((n + 255) / 256)); };

  k_embed<<<grid((long)NN * FEATD), 256, 0, stream>>>(x_idx, pos, res_t, atm_t, cnt_t,
                                                      xyzW, xyzb, hbr[0], hbr[1], hbr[2]);
  k_edges<<<grid(E_ALL), 256, 0, stream>>>(ei, ri, esrc, edst);

  for (int L = 0; L < 3; ++L) {
    int eoff = (L == 0) ? 0 : E_OFF1;
    int ecnt = (L == 0) ? E_ALL : E_LATE;
    for (int br = 0; br < 3; ++br) {                 // 0=C, 1=B, 2=A
      const float* W  = Wb_[br] + (size_t)L * FEATD * NCOLS;
      const float* aL = ab_[br] + (size_t)L * 2 * 2 * 128;
      const float* bL = bb_[br] + (size_t)L * FEATD;
      int act = (br < 2) ? 1 : 0;                    // A branch: no activation
      k_cvt  <<<grid((long)NN * FEATD), 256, 0, stream>>>(hbr[br], hbf, NN * FEATD);
      k_wprep<<<grid((long)NCOLS * FEATD), 256, 0, stream>>>(W, Wt, NCOLS);
      long waves = (long)(NN / 16) * ((NCOLS / 16) / 2);      // 2 col tiles per wave
      k_gemm<NCOLS, 2><<<grid(waves * 32), 256, 0, stream>>>(hbf, Wt, hh, NN);
      k_att  <<<grid((long)NN * 32), 256, 0, stream>>>(hh, aL, als, ald);
      k_init <<<grid((long)NN * FEATD), 256, 0, stream>>>(outb, mu, den);
      k_p1   <<<grid(ecnt), 256, 0, stream>>>(esrc + eoff, edst + eoff, als, ald, mu, ecnt);
      k_p2   <<<grid(ecnt), 256, 0, stream>>>(esrc + eoff, edst + eoff, als, ald, mu, den, exb, ecnt);
      k_p3   <<<grid((long)ecnt * 4), 256, 0, stream>>>(esrc + eoff, edst + eoff, exb, den, hh, outb, ecnt);
      k_bact <<<grid((long)NN * FEATD), 256, 0, stream>>>(outb, bL, hbr[br], act);
    }
  }

  // output projections [N,128] @ [128,16] (+bias, hard_swish on C only) via WMMA
  for (int br = 0; br < 3; ++br) {
    k_cvt  <<<grid((long)NN * FEATD), 256, 0, stream>>>(hbr[br], hbf, NN * FEATD);
    k_wprep<<<grid((long)CALD * FEATD), 256, 0, stream>>>(Wo_[br], Wt, CALD);
    long waves = (long)(NN / 16);                              // single col tile
    k_gemm<CALD, 1><<<grid(waves * 32), 256, 0, stream>>>(hbf, Wt, fin[br], NN);
    k_obact<<<grid((long)NN * CALD), 256, 0, stream>>>(fin[br], bo_[br], br == 0 ? 1 : 0);
  }

  k_zero8 <<<1, 32, 0, stream>>>(out);
  k_energy<<<grid(E_RAD), 256, 0, stream>>>(ri, pos, batch, fin[0], fin[1], fin[2], out);
  (void)in_sizes; (void)n_in; (void)out_size; (void)ws_size;
}
